// Attention_17471926960981
// MI455X (gfx1250) — matrix operand order repaired and checked by reading, not compiled, not run on hardware
//
#include <hip/hip_runtime.h>
#include <hip/hip_bf16.h>

typedef __attribute__((ext_vector_type(16))) __bf16 v16bf;
typedef __attribute__((ext_vector_type(8)))  __bf16 v8bf;
typedef __attribute__((ext_vector_type(8)))  float  v8f;
typedef __attribute__((ext_vector_type(4)))  float  v4f;
typedef __attribute__((ext_vector_type(4)))  int    v4i;

union AFrag { v16bf v; v8bf h[2]; };

// ---------------- async global->LDS (CDNA5, ASYNCcnt) ----------------
#if __has_builtin(__builtin_amdgcn_global_load_async_to_lds_b128) && \
    __has_builtin(__builtin_amdgcn_s_wait_asynccnt)
#define USE_ASYNC_LDS 1
#else
#define USE_ASYNC_LDS 0
#endif

__device__ __forceinline__ void async_copy_b128(const __bf16* g, __bf16* l) {
#if USE_ASYNC_LDS
    __builtin_amdgcn_global_load_async_to_lds_b128(
        (__attribute__((address_space(1))) v4i*)g,
        (__attribute__((address_space(3))) v4i*)l, 0, 0);
#else
    *(v8bf*)l = *(const v8bf*)g;   // synchronous fallback
#endif
}
__device__ __forceinline__ void async_wait0() {
#if USE_ASYNC_LDS
    __builtin_amdgcn_s_wait_asynccnt(0);
#endif
}

// ---------- weight transpose + fp32 -> bf16 ----------
__global__ __launch_bounds__(256)
void wtrans_kernel(const float* __restrict__ W, __bf16* __restrict__ Wt, int D) {
    int idx = blockIdx.x * 256 + threadIdx.x;
    int n = idx % D;
    int k = idx / D;
    if (k < D) Wt[(size_t)n * D + k] = (__bf16)W[(size_t)k * D + n];
}

// ---------- projection GEMM (fp32 A -> bf16 frags), 2x4 register tiling ----------
// C[M,N] = A[M,K] @ Bsrc^T + bias ; Bsrc is [N,K] row-major (pre-transposed weights)
// TRANS: write output as Cbf[(m/Sdim)*N*Sdim + n*Sdim + (m%Sdim)]  (for V)
template<bool TRANS>
__global__ __launch_bounds__(256)
void proj_gemm(const float* __restrict__ Af, const __bf16* __restrict__ Bsrc,
               __bf16* __restrict__ Cbf, const float* __restrict__ bias,
               int M, int N, int K, float alpha, int Sdim)
{
    const int lane = threadIdx.x & 31;
    const int wave = threadIdx.x >> 5;
    const int tilesN = N >> 6;
    const long tile  = (long)blockIdx.x * 8 + wave;
    const long tiles = (long)(M >> 5) * tilesN;
    if (tile >= tiles) return;
    const int m0 = (int)(tile / tilesN) << 5;
    const int n0 = (int)(tile % tilesN) << 6;
    const int l  = lane & 15;
    const int hi = lane >> 4;

    v8f acc[2][4] = {};

    for (int k0 = 0; k0 < K; k0 += 32) {
        const int kA = k0 + hi * 16;
        v16bf a[2];
        #pragma unroll
        for (int mi = 0; mi < 2; ++mi) {
            const float* ap = Af + (size_t)(m0 + 16 * mi + l) * K + kA;
            v4f f0 = *(const v4f*)(ap);
            v4f f1 = *(const v4f*)(ap + 4);
            v4f f2 = *(const v4f*)(ap + 8);
            v4f f3 = *(const v4f*)(ap + 12);
            v16bf t;
            #pragma unroll
            for (int i = 0; i < 4; ++i) {
                t[i]      = (__bf16)f0[i];
                t[4 + i]  = (__bf16)f1[i];
                t[8 + i]  = (__bf16)f2[i];
                t[12 + i] = (__bf16)f3[i];
            }
            a[mi] = t;
        }
        v16bf b[4];
        #pragma unroll
        for (int ni = 0; ni < 4; ++ni) {
            const __bf16* bp = Bsrc + (size_t)(n0 + 16 * ni + l) * K + k0 + hi * 16;
            b[ni] = *(const v16bf*)bp;
        }
        #pragma unroll
        for (int mi = 0; mi < 2; ++mi)
            #pragma unroll
            for (int ni = 0; ni < 4; ++ni)
                acc[mi][ni] = __builtin_amdgcn_wmma_f32_16x16x32_bf16(
                                  false, a[mi], false, b[ni], (short)0,
                                  acc[mi][ni], false, false);
    }

    #pragma unroll
    for (int ni = 0; ni < 4; ++ni) {
        const int n = n0 + 16 * ni + l;
        const float bvv = bias[n];
        #pragma unroll
        for (int mi = 0; mi < 2; ++mi)
            #pragma unroll
            for (int r = 0; r < 8; ++r) {
                const int m = m0 + 16 * mi + r + hi * 8;
                const float v = acc[mi][ni][r] * alpha + bvv;
                if (TRANS) {
                    const int bb = m / Sdim, ss = m % Sdim;
                    Cbf[((size_t)bb * N + n) * Sdim + ss] = (__bf16)v;
                } else {
                    Cbf[(size_t)m * N + n] = (__bf16)v;
                }
            }
    }
}

// ---------- LDS-staged, double-buffered bf16 GEMM (async copies) ----------
// Block = 8 waves (2 M x 4 N), block tile 64x256, wave tile 32x64, K step 32.
// C[M,N] = alpha * A[M,K] @ Bsrc^T ; A bf16 [M,K] (lda=K), Bsrc bf16 [N,K] (ldb=K).
#define KSTEP 32
#define LPAD  8
#define LROW  (KSTEP + LPAD)   // 40 bf16 = 80B rows, 16B-aligned chunks

__global__ __launch_bounds__(256)
void gemm_bf16_lds(const __bf16* __restrict__ Ab, const __bf16* __restrict__ Bs,
                   float* __restrict__ Cf, int M, int N, int K, float alpha)
{
    __shared__ __bf16 Asm[2][64][LROW];    // 10 KB
    __shared__ __bf16 Bsm[2][256][LROW];   // 40 KB

    const int lane = threadIdx.x & 31;
    const int wave = threadIdx.x >> 5;
    const int wm = wave >> 2;              // 0..1
    const int wn = wave & 3;               // 0..3
    const int nb = N >> 8;                 // blocks along N (256 wide)
    const int m_blk = (blockIdx.x / nb) << 6;
    const int n_blk = (blockIdx.x % nb) << 8;
    const int l  = lane & 15;
    const int hi = lane >> 4;

    // per-thread staging slots: one 16B chunk of A, four of B
    const int ar = threadIdx.x >> 2;            // 0..63
    const int ac = (threadIdx.x & 3) * 8;       // 0,8,16,24

    auto stage = [&](int buf, int k0) {
        async_copy_b128(Ab + (size_t)(m_blk + ar) * K + k0 + ac, &Asm[buf][ar][ac]);
        #pragma unroll
        for (int j = 0; j < 4; ++j) {
            const int idx = threadIdx.x + j * 256;
            const int br = idx >> 2;            // 0..255
            const int bc = (idx & 3) * 8;
            async_copy_b128(Bs + (size_t)(n_blk + br) * K + k0 + bc, &Bsm[buf][br][bc]);
        }
    };

    v8f acc[2][4] = {};

    stage(0, 0);
    async_wait0();
    __syncthreads();

    for (int k0 = 0; k0 < K; k0 += KSTEP) {
        const int cur = (k0 / KSTEP) & 1;
        const bool more = (k0 + KSTEP) < K;
        if (more) stage(cur ^ 1, k0 + KSTEP);

        // A fragments from LDS
        v16bf a[2];
        #pragma unroll
        for (int mi = 0; mi < 2; ++mi) {
            const int row = wm * 32 + 16 * mi + l;
            AFrag u;
            u.h[0] = *(const v8bf*)&Asm[cur][row][hi * 8];
            u.h[1] = *(const v8bf*)&Asm[cur][row][hi * 8 + 16];
            a[mi] = u.v;
        }
        // B fragments from LDS
        v16bf b[4];
        #pragma unroll
        for (int ni = 0; ni < 4; ++ni) {
            const int row = wn * 64 + 16 * ni + l;
            AFrag u;
            u.h[0] = *(const v8bf*)&Bsm[cur][row][hi * 8];
            u.h[1] = *(const v8bf*)&Bsm[cur][row][hi * 8 + 16];
            b[ni] = u.v;
        }
        #pragma unroll
        for (int mi = 0; mi < 2; ++mi)
            #pragma unroll
            for (int ni = 0; ni < 4; ++ni)
                acc[mi][ni] = __builtin_amdgcn_wmma_f32_16x16x32_bf16(
                                  false, a[mi], false, b[ni], (short)0,
                                  acc[mi][ni], false, false);

        if (more) async_wait0();
        __syncthreads();
    }

    #pragma unroll
    for (int ni = 0; ni < 4; ++ni) {
        const int n = n_blk + wn * 64 + 16 * ni + l;
        #pragma unroll
        for (int mi = 0; mi < 2; ++mi)
            #pragma unroll
            for (int r = 0; r < 8; ++r) {
                const int m = m_blk + wm * 32 + 16 * mi + r + hi * 8;
                Cf[(size_t)m * N + n] = acc[mi][ni][r] * alpha;
            }
    }
}

// ---------- row softmax: one 256-thread block per row, write bf16 probs ----------
__global__ __launch_bounds__(256)
void softmax_rows(const float* __restrict__ Sc, __bf16* __restrict__ P, int S) {
    __shared__ float red[256];
    const int row = blockIdx.x;
    const int t = threadIdx.x;
    const float* r = Sc + (size_t)row * S;

    float lmax = -3.4e38f;
    for (int j = t; j < S; j += 256) lmax = fmaxf(lmax, r[j]);
    red[t] = lmax; __syncthreads();
    for (int off = 128; off > 0; off >>= 1) {
        if (t < off) red[t] = fmaxf(red[t], red[t + off]);
        __syncthreads();
    }
    const float m = red[0]; __syncthreads();

    float lsum = 0.0f;
    for (int j = t; j < S; j += 256) lsum += __expf(r[j] - m);
    red[t] = lsum; __syncthreads();
    for (int off = 128; off > 0; off >>= 1) {
        if (t < off) red[t] += red[t + off];
        __syncthreads();
    }
    const float inv = 1.0f / red[0];

    __bf16* p = P + (size_t)row * S;
    for (int j = t; j < S; j += 256) p[j] = (__bf16)(__expf(r[j] - m) * inv);
}

// ---------- host launcher ----------
extern "C" void kernel_launch(void* const* d_in, const int* in_sizes, int n_in,
                              void* d_out, int out_size, void* d_ws, size_t ws_size,
                              hipStream_t stream)
{
    (void)in_sizes; (void)n_in; (void)out_size; (void)ws_size;
    const int Bb = 4, S = 2048, D = 1024;

    const float* q  = (const float*)d_in[0];
    const float* k  = (const float*)d_in[1];
    const float* v  = (const float*)d_in[2];
    const float* Wq = (const float*)d_in[3];
    const float* bq = (const float*)d_in[4];
    const float* Wk = (const float*)d_in[5];
    const float* bk = (const float*)d_in[6];
    const float* Wv = (const float*)d_in[7];
    const float* bv = (const float*)d_in[8];
    float* out = (float*)d_out;

    char* ws = (char*)d_ws;
    size_t off = 0;
    auto carve = [&](size_t bytes) -> void* {
        void* p = ws + off;
        off = (off + bytes + 255) & ~(size_t)255;
        return p;
    };
    __bf16* Qbf = (__bf16*)carve((size_t)Bb * S * D * 2);  // [B,S,D] bf16
    __bf16* Kbf = (__bf16*)carve((size_t)Bb * S * D * 2);  // [B,S,D] bf16
    __bf16* Vtb = (__bf16*)carve((size_t)Bb * D * S * 2);  // [B,D,S] bf16 (transposed)
    __bf16* Wqt = (__bf16*)carve((size_t)D * D * 2);
    __bf16* Wkt = (__bf16*)carve((size_t)D * D * 2);
    __bf16* Wvt = (__bf16*)carve((size_t)D * D * 2);
    float*  Sc  = (float*) carve((size_t)S * S * 4);       // one batch of scores
    __bf16* Pb  = (__bf16*)carve((size_t)S * S * 2);       // one batch of probs

    // 1) weights -> transposed bf16
    const int tgrid = (D * D) / 256;
    wtrans_kernel<<<tgrid, 256, 0, stream>>>(Wq, Wqt, D);
    wtrans_kernel<<<tgrid, 256, 0, stream>>>(Wk, Wkt, D);
    wtrans_kernel<<<tgrid, 256, 0, stream>>>(Wv, Wvt, D);

    // 2) projections: [B*S, D] = in @ W + b
    const int M = Bb * S;
    const long ptiles = (long)(M / 32) * (D / 64);
    const int pblocks = (int)((ptiles + 7) / 8);
    proj_gemm<false><<<pblocks, 256, 0, stream>>>(q, Wqt, Qbf, bq, M, D, D, 1.0f, S);
    proj_gemm<false><<<pblocks, 256, 0, stream>>>(k, Wkt, Kbf, bk, M, D, D, 1.0f, S);
    proj_gemm<true><<<pblocks, 256, 0, stream>>>(v, Wvt, Vtb, bv, M, D, D, 1.0f, S);

    // 3) per-batch attention (reuse score/prob buffers)
    const float inv_sqrt_d = 1.0f / 32.0f;  // sqrt(1024) = 32
    const int sblocks = (S / 64) * (S / 256);   // 256
    const int vblocks = (S / 64) * (D / 256);   // 128
    for (int b = 0; b < Bb; ++b) {
        const __bf16* Qb = Qbf + (size_t)b * S * D;
        const __bf16* Kb = Kbf + (size_t)b * S * D;
        const __bf16* Vb = Vtb + (size_t)b * D * S;
        // scores = (Q @ K^T) / sqrt(D)   [S,S] fp32
        gemm_bf16_lds<<<sblocks, 256, 0, stream>>>(Qb, Kb, Sc, S, S, D, inv_sqrt_d);
        // softmax rows -> bf16 probs
        softmax_rows<<<S, 256, 0, stream>>>(Sc, Pb, S);
        // out = P @ V   (V stored as [D,S] so its rows feed B columns)
        gemm_bf16_lds<<<vblocks, 256, 0, stream>>>(Pb, Vb, out + (size_t)b * S * D,
                                                   S, D, S, 1.0f);
    }
}
